// DistinctionLoss_19344532702281
// MI455X (gfx1250) — matrix operand order repaired and checked by reading, not compiled, not run on hardware
//
#include <hip/hip_runtime.h>
#include <stdint.h>

// ---------------- problem constants (from reference) ----------------
#define NB 16
#define NN 2048
#define ND 256
#define NH 384
#define NW 384
#define NHW (NH*NW)
#define NP (NB*NHW)        // 2359296 pixels total
#define CAP 4096           // per-batch candidate cap (expect ~2304)
#define TOPK 200

// ---------------- CDNA5 WMMA types ----------------
typedef __attribute__((ext_vector_type(16))) __bf16 v16bf;
typedef __attribute__((ext_vector_type(8)))  float  v8f;
typedef __attribute__((ext_vector_type(4)))  unsigned int v4u;
typedef __attribute__((ext_vector_type(2)))  unsigned int v2u;
typedef __attribute__((ext_vector_type(4)))  float  v4f;

union Frag { v4u u[2]; v16bf v; };

// 7-tap gaussian (exp(-0.5*x^2), x=-3..3, normalized)
__device__ __constant__ float GW[7] = {
  0.004433048f, 0.054005583f, 0.242036229f, 0.399050280f,
  0.242036229f, 0.054005583f, 0.004433048f };

// ---------------- helpers ----------------
__device__ __forceinline__ void bf16split(float x, unsigned short& h, unsigned short& l) {
  unsigned u  = __float_as_uint(x);
  unsigned hb = (u + 0x7fffu + ((u >> 16) & 1u)) >> 16;   // RNE to bf16
  float hf = __uint_as_float(hb << 16);
  float lo = x - hf;
  unsigned ul = __float_as_uint(lo);
  unsigned lb = (ul + 0x7fffu + ((ul >> 16) & 1u)) >> 16;
  h = (unsigned short)hb; l = (unsigned short)lb;
}

// CDNA5 async copy: 16B global -> LDS, tracked by ASYNCcnt (ISA 15.18.3 op 98).
// LDS byte offset = low 32 bits of the generic pointer (LDS aperture: addr[31:0]).
__device__ __forceinline__ void async_copy16(unsigned lds_off, const void* g) {
  asm volatile("global_load_async_to_lds_b128 %0, %1, off"
               :: "v"(lds_off), "v"((unsigned long long)(uintptr_t)g)
               : "memory");
}
#define LDS_OFF(p) ((unsigned)(uintptr_t)(const void*)(p))

#if __has_builtin(__builtin_amdgcn_s_wait_asynccnt)
#define WAIT_ASYNC(n) __builtin_amdgcn_s_wait_asynccnt(n)
#else
#define WAIT_ASYNC(n) asm volatile("s_wait_asynccnt %0" :: "n"(n) : "memory")
#endif

// ---------------- zero accumulators / counters ----------------
__global__ void k_zero(double* acc, int* cnt) {
  int t = threadIdx.x;
  if (t < 4)  acc[t] = 0.0;
  if (t < NB) cnt[t] = 0;
}

// ---------------- grayscale ----------------
__global__ void k_gray(const float* __restrict__ img, float* __restrict__ gray) {
  int p = blockIdx.x * 256 + threadIdx.x;
  if (p >= NP) return;
  int b = p / NHW, q = p - b * NHW;
  const float* ib = img + (size_t)b * 3 * NHW;
  gray[p] = 0.299f * ib[q] + 0.587f * ib[NHW + q] + 0.114f * ib[2 * NHW + q];
}

// ---------------- sobel/8 gradients (edge/clamp pad) -> products ----------------
__global__ void k_grad(const float* __restrict__ gray,
                       float* __restrict__ dx2, float* __restrict__ dy2,
                       float* __restrict__ dxy) {
  int p = blockIdx.x * 256 + threadIdx.x;
  if (p >= NP) return;
  int b = p / NHW, q = p - b * NHW;
  int y = q / NW, x = q - y * NW;
  const float* g = gray + (size_t)b * NHW;
  auto G = [&](int yy, int xx) {
    yy = yy < 0 ? 0 : (yy > NH - 1 ? NH - 1 : yy);
    xx = xx < 0 ? 0 : (xx > NW - 1 ? NW - 1 : xx);
    return g[yy * NW + xx];
  };
  float a00 = G(y-1,x-1), a01 = G(y-1,x), a02 = G(y-1,x+1);
  float a10 = G(y, x-1),                  a12 = G(y, x+1);
  float a20 = G(y+1,x-1), a21 = G(y+1,x), a22 = G(y+1,x+1);
  float gx = (a02 - a00 + 2.f * (a12 - a10) + a22 - a20) * 0.125f;
  float gy = (a20 - a00 + 2.f * (a21 - a01) + a22 - a02) * 0.125f;
  dx2[p] = gx * gx; dy2[p] = gy * gy; dxy[p] = gx * gy;
}

// ---------------- 7x7 gaussian (reflect pad) + GFTT min-eigenvalue ----------------
__global__ void k_blur_resp(const float* __restrict__ dx2, const float* __restrict__ dy2,
                            const float* __restrict__ dxy, float* __restrict__ resp) {
  int p = blockIdx.x * 256 + threadIdx.x;
  if (p >= NP) return;
  int b = p / NHW, q = p - b * NHW;
  int y = q / NW, x = q - y * NW;
  size_t base = (size_t)b * NHW;
  float s0 = 0.f, s1 = 0.f, s2 = 0.f;
  #pragma unroll
  for (int wy = 0; wy < 7; ++wy) {
    int iy = y + wy - 3;
    iy = iy < 0 ? -iy : (iy >= NH ? 2 * NH - 2 - iy : iy);   // np 'reflect'
    float wv = GW[wy];
    size_t ro = base + (size_t)iy * NW;
    #pragma unroll
    for (int wx = 0; wx < 7; ++wx) {
      int ix = x + wx - 3;
      ix = ix < 0 ? -ix : (ix >= NW ? 2 * NW - 2 - ix : ix);
      float w = wv * GW[wx];
      s0 += w * dx2[ro + ix]; s1 += w * dy2[ro + ix]; s2 += w * dxy[ro + ix];
    }
  }
  float tr = s0 + s1;
  float det = s0 * s1 - s2 * s2;
  resp[p] = 0.5f * (tr - sqrtf(fabsf(tr * tr - 4.f * det)));
}

// ---------------- 5x5 NMS (keep == windowed max) ----------------
__global__ void k_nms(const float* __restrict__ resp, float* __restrict__ out) {
  int p = blockIdx.x * 256 + threadIdx.x;
  if (p >= NP) return;
  int b = p / NHW, q = p - b * NHW;
  int y = q / NW, x = q - y * NW;
  const float* r = resp + (size_t)b * NHW;
  float v = r[q], m = -3.4e38f;
  for (int dy = -2; dy <= 2; ++dy) {
    int iy = y + dy; if (iy < 0 || iy >= NH) continue;
    for (int dx = -2; dx <= 2; ++dx) {
      int ix = x + dx; if (ix < 0 || ix >= NW) continue;
      m = fmaxf(m, r[iy * NW + ix]);
    }
  }
  out[p] = (v == m) ? v : 0.f;
}

// ---------------- per 8x8 block: keep == block max, >0 -> candidate ----------------
__global__ void k_blk(const float* __restrict__ nmsv, float* __restrict__ cval,
                      int* __restrict__ cidx, int* __restrict__ cnt) {
  __shared__ float sm[64];
  int t = threadIdx.x, blk = blockIdx.x;
  int b = blk / (48 * 48), r2 = blk - b * 48 * 48;
  int by = r2 / 48, bx = r2 - by * 48;
  int y = by * 8 + (t >> 3), x = bx * 8 + (t & 7);
  int q = y * NW + x;
  float v = nmsv[(size_t)b * NHW + q];
  sm[t] = v; __syncthreads();
  for (int s = 32; s > 0; s >>= 1) {
    if (t < s) sm[t] = fmaxf(sm[t], sm[t + s]);
    __syncthreads();
  }
  float bmax = sm[0];
  if (v > 0.f && v == bmax) {
    int pos = atomicAdd(&cnt[b], 1);
    if (pos < CAP) { cval[b * CAP + pos] = v; cidx[b * CAP + pos] = q; }
  }
}

// ---------------- per batch: bitonic sort candidates, top-200, gather sum ----------------
__global__ __launch_bounds__(1024) void k_topk(const float* __restrict__ cval,
                                               const int* __restrict__ cidx,
                                               const int* __restrict__ cnt,
                                               const float* __restrict__ sdense,
                                               double* acc) {
  __shared__ float tv[CAP];
  __shared__ int   ti[CAP];
  int b = blockIdx.x, t = threadIdx.x;
  int n = cnt[b]; if (n > CAP) n = CAP;
  for (int i = t; i < CAP; i += 1024) {
    if (i < n) { tv[i] = cval[b * CAP + i]; ti[i] = cidx[b * CAP + i]; }
    else       { tv[i] = -1.f; ti[i] = -1; }
  }
  __syncthreads();
  for (int k = 2; k <= CAP; k <<= 1)
    for (int j = k >> 1; j > 0; j >>= 1) {
      for (int i = t; i < CAP; i += 1024) {
        int ixj = i ^ j;
        if (ixj > i) {
          bool desc = ((i & k) == 0);
          float a = tv[i], bb = tv[ixj];
          if (desc ? (a < bb) : (a > bb)) {
            tv[i] = bb; tv[ixj] = a;
            int tmp = ti[i]; ti[i] = ti[ixj]; ti[ixj] = tmp;
          }
        }
      }
      __syncthreads();
    }
  if (t < TOPK && tv[t] > 0.f)
    atomicAdd(&acc[0], -(double)sdense[(size_t)b * NHW + ti[t]]);
}

// ---------------- sum softplus(scores_dense) ----------------
__global__ void k_sp(const float* __restrict__ sd, double* acc) {
  __shared__ float sr[256];
  int p = blockIdx.x * 256 + threadIdx.x;
  float v = 0.f;
  if (p < NP) {
    float x = sd[p];
    v = fmaxf(x, 0.f) + log1pf(expf(-fabsf(x)));
  }
  sr[threadIdx.x] = v; __syncthreads();
  for (int s = 128; s > 0; s >>= 1) {
    if (threadIdx.x < s) sr[threadIdx.x] += sr[threadIdx.x + s];
    __syncthreads();
  }
  if (threadIdx.x == 0) atomicAdd(&acc[0], (double)sr[0]);
}

// ---------------- row norms of descriptors (one wave32 per row) ----------------
__global__ void k_norms(const float* __restrict__ desc, float* __restrict__ norms) {
  int row  = blockIdx.x * 8 + (threadIdx.x >> 5);
  int lane = threadIdx.x & 31;
  const float* r = desc + (size_t)row * ND;
  float s = 0.f;
  #pragma unroll
  for (int q = 0; q < 8; ++q) { float x = r[lane + 32 * q]; s += x * x; }
  #pragma unroll
  for (int m = 16; m > 0; m >>= 1) s += __shfl_xor(s, m, 32);
  if (lane == 0) norms[row] = sqrtf(s);
}

// ---------------- one-shot fp32 -> bf16 hi/lo split of descriptors ----------------
__global__ void k_split(const float* __restrict__ x, unsigned short* __restrict__ hi,
                        unsigned short* __restrict__ lo) {
  int p = blockIdx.x * 256 + threadIdx.x;     // quad index, total NB*NN*ND/4
  v4f f = *(const v4f*)(x + (size_t)p * 4);
  unsigned short h[4], l[4];
  #pragma unroll
  for (int e = 0; e < 4; ++e) bf16split(f[e], h[e], l[e]);
  v2u H, L;
  H[0] = (unsigned)h[0] | ((unsigned)h[1] << 16); H[1] = (unsigned)h[2] | ((unsigned)h[3] << 16);
  L[0] = (unsigned)l[0] | ((unsigned)l[1] << 16); L[1] = (unsigned)l[2] | ((unsigned)l[3] << 16);
  *(v2u*)(hi + (size_t)p * 4) = H;
  *(v2u*)(lo + (size_t)p * 4) = L;
}

// ---------------- batched Gram GEMM: async double-buffered bf16 hi/lo WMMA ----------------
#define LDA 40   // ushort stride: 80B rows -> every 16B chunk aligned, conflict-free A frags
__global__ __launch_bounds__(256) void k_gemm(const unsigned short* __restrict__ dHi,
                                              const unsigned short* __restrict__ dLo,
                                              const float* __restrict__ norms,
                                              double* acc) {
  __shared__ __align__(16) unsigned short sAh[2][128 * LDA];
  __shared__ __align__(16) unsigned short sAl[2][128 * LDA];
  __shared__ __align__(16) unsigned short sBh[2][64 * LDA];
  __shared__ __align__(16) unsigned short sBl[2][64 * LDA];
  __shared__ float sred[256];

  int blk = blockIdx.x;
  int b  = blk >> 9;           // 16*32 = 512 tiles per batch
  int r  = blk & 511;
  int tm = r >> 5, tn = r & 31;
  int m0 = tm * 128, n0 = tn * 64;
  int tid = threadIdx.x, lane = tid & 31, w = tid >> 5;
  int m_off = (w & 3) * 32, n_off = (w >> 2) * 32;
  int half = lane >> 4, l15 = lane & 15;
  const unsigned short* Xh = dHi + (size_t)b * NN * ND;
  const unsigned short* Xl = dLo + (size_t)b * NN * ND;

  // per-thread copy assignment: 6x 16B chunks per K-chunk buffer
  int rA = tid >> 2;            // 0..63
  int cc = (tid & 3) * 8;       // ushort column offset {0,8,16,24}
  int rA2 = rA + 64;            // second A row group

  auto issue = [&](int buf, int k0) {
    async_copy16(LDS_OFF(&sAh[buf][rA  * LDA + cc]), Xh + (size_t)(m0 + rA ) * ND + k0 + cc);
    async_copy16(LDS_OFF(&sAh[buf][rA2 * LDA + cc]), Xh + (size_t)(m0 + rA2) * ND + k0 + cc);
    async_copy16(LDS_OFF(&sAl[buf][rA  * LDA + cc]), Xl + (size_t)(m0 + rA ) * ND + k0 + cc);
    async_copy16(LDS_OFF(&sAl[buf][rA2 * LDA + cc]), Xl + (size_t)(m0 + rA2) * ND + k0 + cc);
    async_copy16(LDS_OFF(&sBh[buf][rA  * LDA + cc]), Xh + (size_t)(n0 + rA ) * ND + k0 + cc);
    async_copy16(LDS_OFF(&sBl[buf][rA  * LDA + cc]), Xl + (size_t)(n0 + rA ) * ND + k0 + cc);
  };

  v8f accv[2][2];
  for (int a = 0; a < 2; ++a)
    for (int c = 0; c < 2; ++c)
      for (int e = 0; e < 8; ++e) accv[a][c][e] = 0.f;

  issue(0, 0);
  for (int kc = 0; kc < 8; ++kc) {
    int cur = kc & 1;
    if (kc < 7) { issue(cur ^ 1, (kc + 1) * 32); WAIT_ASYNC(6); }
    else        { WAIT_ASYNC(0); }
    __syncthreads();   // all waves' copies for buf[cur] retired

    // A frag: lane row = l15; elems 0..7 -> K=8h+e, 8..15 -> K=16+8h+(e-8)
    Frag ah[2], al[2], bh[2], bl[2];
    #pragma unroll
    for (int mt = 0; mt < 2; ++mt) {
      const unsigned short* ph = &sAh[cur][(m_off + mt * 16 + l15) * LDA];
      const unsigned short* pl = &sAl[cur][(m_off + mt * 16 + l15) * LDA];
      ah[mt].u[0] = *(const v4u*)(ph + 8 * half);
      ah[mt].u[1] = *(const v4u*)(ph + 16 + 8 * half);
      al[mt].u[0] = *(const v4u*)(pl + 8 * half);
      al[mt].u[1] = *(const v4u*)(pl + 16 + 8 * half);
    }
    // B frag: lane fixed N = l15; elems 0..7 -> K=8h+e, 8..15 -> K=16+8h+(e-8)
    #pragma unroll
    for (int nt = 0; nt < 2; ++nt) {
      const unsigned short* ph = &sBh[cur][(n_off + nt * 16 + l15) * LDA];
      const unsigned short* pl = &sBl[cur][(n_off + nt * 16 + l15) * LDA];
      bh[nt].u[0] = *(const v4u*)(ph + 8 * half);
      bh[nt].u[1] = *(const v4u*)(ph + 16 + 8 * half);
      bl[nt].u[0] = *(const v4u*)(pl + 8 * half);
      bl[nt].u[1] = *(const v4u*)(pl + 16 + 8 * half);
    }
    // dot = hi*hi + hi*lo + lo*hi  (all fp32-accumulated)
    #pragma unroll
    for (int mt = 0; mt < 2; ++mt)
      #pragma unroll
      for (int nt = 0; nt < 2; ++nt) {
        accv[mt][nt] = __builtin_amdgcn_wmma_f32_16x16x32_bf16(
            false, ah[mt].v, false, bh[nt].v, (short)0, accv[mt][nt], false, false);
        accv[mt][nt] = __builtin_amdgcn_wmma_f32_16x16x32_bf16(
            false, ah[mt].v, false, bl[nt].v, (short)0, accv[mt][nt], false, false);
        accv[mt][nt] = __builtin_amdgcn_wmma_f32_16x16x32_bf16(
            false, al[mt].v, false, bh[nt].v, (short)0, accv[mt][nt], false, false);
      }
    __syncthreads();   // protect buf[cur] before it is overwritten at kc+1's issue
  }

  // epilogue: relu(dot / max(ni*nj, eps)), reduce, atomic accumulate
  const float* nb = norms + b * NN;
  float s = 0.f;
  #pragma unroll
  for (int mt = 0; mt < 2; ++mt) {
    int ibase = m0 + m_off + mt * 16 + 8 * half;   // C/D: M = r + 8*half
    #pragma unroll
    for (int nt = 0; nt < 2; ++nt) {
      int j = n0 + n_off + nt * 16 + l15;          // C/D: N = lane&15
      float nj = nb[j];
      #pragma unroll
      for (int rr = 0; rr < 8; ++rr) {
        float pc = accv[mt][nt][rr] / fmaxf(nb[ibase + rr] * nj, 1e-8f);
        s += fmaxf(pc, 0.f);
      }
    }
  }
  sred[tid] = s; __syncthreads();
  for (int st = 128; st > 0; st >>= 1) {
    if (tid < st) sred[tid] += sred[tid + st];
    __syncthreads();
  }
  if (tid == 0) atomicAdd(&acc[1], (double)sred[0]);
}

// ---------------- combine: mean BCE + mean relu(pcos) ----------------
__global__ void k_final(const double* acc, float* out) {
  out[0] = (float)(acc[0] / (double)NP + acc[1] / ((double)NB * NN * NN));
}

// ---------------- launcher ----------------
extern "C" void kernel_launch(void* const* d_in, const int* in_sizes, int n_in,
                              void* d_out, int out_size, void* d_ws, size_t ws_size,
                              hipStream_t stream) {
  (void)in_sizes; (void)n_in; (void)out_size; (void)ws_size;
  const float* desc   = (const float*)d_in[0];   // [16,2048,256]
  const float* sdense = (const float*)d_in[2];   // [16,1,384,384]
  const float* imgs   = (const float*)d_in[3];   // [16,3,384,384]
  float* out = (float*)d_out;

  char* ws = (char*)d_ws;
  double* acc = (double*)ws;                 // acc[0]=bce numerator, acc[1]=pcos sum
  int* cnt = (int*)(ws + 32);                // 16 counters
  float* base = (float*)(ws + 128);
  float* gray = base;                        // plane (also resp)
  float* p0 = base + (size_t)NP;             // dx2, later nms result
  float* p1 = base + (size_t)2 * NP;         // dy2
  float* p2 = base + (size_t)3 * NP;         // dxy
  float* norms = base + (size_t)4 * NP;      // 32768 floats
  float* cval = norms + (size_t)NB * NN;     // 16*4096 floats
  int* cidx = (int*)(cval + (size_t)NB * CAP);
  unsigned short* dHi = (unsigned short*)(cidx + (size_t)NB * CAP);  // 16B-aligned
  unsigned short* dLo = dHi + (size_t)NB * NN * ND;

  int gp = (NP + 255) / 256;
  k_zero<<<1, 64, 0, stream>>>(acc, cnt);
  k_gray<<<gp, 256, 0, stream>>>(imgs, gray);
  k_grad<<<gp, 256, 0, stream>>>(gray, p0, p1, p2);
  k_blur_resp<<<gp, 256, 0, stream>>>(p0, p1, p2, gray);      // resp -> gray plane
  k_nms<<<gp, 256, 0, stream>>>(gray, p0);                    // nms -> p0
  k_blk<<<NB * 48 * 48, 64, 0, stream>>>(p0, cval, cidx, cnt);
  k_topk<<<NB, 1024, 0, stream>>>(cval, cidx, cnt, sdense, acc);
  k_sp<<<gp, 256, 0, stream>>>(sdense, acc);
  k_norms<<<(NB * NN) / 8, 256, 0, stream>>>(desc, norms);
  k_split<<<(NB * NN * ND / 4) / 256, 256, 0, stream>>>(desc, dHi, dLo);
  k_gemm<<<NB * 16 * 32, 256, 0, stream>>>(dHi, dLo, norms, acc);
  k_final<<<1, 1, 0, stream>>>(acc, out);
}
